// CandidateSubgraphTFModel_88545045775055
// MI455X (gfx1250) — matrix operand order repaired and checked by reading, not compiled, not run on hardware
//
#include <hip/hip_runtime.h>
#include <hip/hip_bf16.h>

// ---------------- problem constants (match reference) ----------------
constexpr int D_  = 512;
constexpr int B_  = 32;
constexpr int S_  = 1024;
constexpr int N_  = 2048;
constexpr int BS_ = B_ * S_;   // 32768
constexpr int ZPAD_ = 1024;    // zero page elements (covers D_ worth of A-frag reads)

typedef __attribute__((ext_vector_type(16))) __bf16 v16b;
typedef __attribute__((ext_vector_type(8)))  __bf16 v8b;
typedef __attribute__((ext_vector_type(4)))  __bf16 v4b;
typedef __attribute__((ext_vector_type(8)))  float  v8f;

struct BPair { v8b lo; v8b hi; };
union  AFrag { v16b v; BPair s; };

// A fragment: lane holds row m=l&15; half-wave selects k offset:
//   elems 0..7  = A[m][k0+ko .. +7]      (ko = (l>=16)?16:0)
//   elems 8..15 = A[m][k0+ko+8 .. +15]
__device__ __forceinline__ v16b load_a(const __bf16* p) {
  AFrag u;
  u.s.lo = *(const v8b*)(p);
  u.s.hi = *(const v8b*)(p + 8);
  return u.v;
}

__device__ __forceinline__ v8f wmma_bf16(v16b a, v16b b, v8f c) {
  return __builtin_amdgcn_wmma_f32_16x16x32_bf16(
      /*neg_a=*/false, a, /*neg_b=*/false, b,
      /*c_mod=*/(short)0, c, /*reuse_a=*/false, /*reuse_b=*/false);
}

// =====================================================================
// Generic bf16 GEMM:  C[m][n] = op( sum_k A[m][k]*Bt[n][k] + bias[n] )
// A: [M,Kd] row-major (lda=Kd).  Bt: [Ncols,Kd] row-major (ldb=Kd).
// Wave tile 32x64, block = 4 waves -> 32x256. Requires M%32==0,
// Ncols%256==0, Kd%32==0 (true for every call below).
// =====================================================================
template <typename OutT, bool RELU, bool TRANS>
__global__ __launch_bounds__(128) void gemm_k(
    const __bf16* __restrict__ A, const __bf16* __restrict__ Bt,
    OutT* __restrict__ C, const float* __restrict__ bias,
    int Kd, int ldc)
{
  const int lane = threadIdx.x;
  const int m0   = blockIdx.y * 32;
  const int n0   = blockIdx.x * 256 + threadIdx.y * 64;
  const int ml   = lane & 15;
  const int hi   = lane >> 4;
  const int ko   = hi * 16;
  const int kb   = hi * 16;

  v8f acc[2][4];
#pragma unroll
  for (int i = 0; i < 2; ++i)
#pragma unroll
    for (int j = 0; j < 4; ++j)
#pragma unroll
      for (int r = 0; r < 8; ++r) acc[i][j][r] = 0.0f;

  const __bf16* a0 = A + (size_t)(m0 + ml)      * Kd + ko;
  const __bf16* a1 = A + (size_t)(m0 + 16 + ml) * Kd + ko;
  const __bf16* b0 = Bt + (size_t)(n0 + ml)     * Kd + kb;

  for (int k0 = 0; k0 < Kd; k0 += 32) {
    v16b af0 = load_a(a0 + k0);
    v16b af1 = load_a(a1 + k0);
#pragma unroll
    for (int j = 0; j < 4; ++j) {
      v16b bf = *(const v16b*)(b0 + (size_t)j * 16 * Kd + k0);
      acc[0][j] = wmma_bf16(af0, bf, acc[0][j]);
      acc[1][j] = wmma_bf16(af1, bf, acc[1][j]);
    }
  }

#pragma unroll
  for (int j = 0; j < 4; ++j) {
    const int c  = n0 + j * 16 + ml;
    const float bv = bias ? bias[c] : 0.0f;
#pragma unroll
    for (int i = 0; i < 2; ++i)
#pragma unroll
      for (int r = 0; r < 8; ++r) {
        const int mr = m0 + i * 16 + hi * 8 + r;
        float v = acc[i][j][r] + bv;
        if (RELU) v = fmaxf(v, 0.0f);
        if (TRANS) C[(size_t)c * ldc + mr] = (OutT)v;
        else       C[(size_t)mr * ldc + c] = (OutT)v;
      }
  }
}

// =====================================================================
// Causal conv as 3 shifted-row GEMM taps over K=D.
// y[b,s,o] = sum_t sum_i x[b, s+t-2, i] * Wt[t][o][i]   (zero for s+t-2<0)
// Out-of-range rows read a zeroed scratch page: the validity predicate is
// folded into a per-lane pointer select OUTSIDE the K-loop, so the inner
// loop is a branch-free load_b128 -> v_wmma stream.
// Tiles of 32 rows never straddle a batch (32 | S).
// =====================================================================
template <bool RELU, bool RES, typename OutT>
__global__ __launch_bounds__(128) void conv_k(
    const __bf16* __restrict__ A, const __bf16* __restrict__ Wt,
    OutT* __restrict__ C, const float* __restrict__ bias,
    const __bf16* __restrict__ Res, const __bf16* __restrict__ Zpage)
{
  const int lane = threadIdx.x;
  const int m0   = blockIdx.y * 32;
  const int n0   = blockIdx.x * 256 + threadIdx.y * 64;
  const int ml   = lane & 15;
  const int hi   = lane >> 4;
  const int ko   = hi * 16;
  const int kb   = hi * 16;
  const int sbase = m0 & (S_ - 1);

  v8f acc[2][4];
#pragma unroll
  for (int i = 0; i < 2; ++i)
#pragma unroll
    for (int j = 0; j < 4; ++j)
#pragma unroll
      for (int r = 0; r < 8; ++r) acc[i][j][r] = 0.0f;

#pragma unroll
  for (int t = 0; t < 3; ++t) {
    const int sh = t - 2;
    // per-lane pointer select (v_cndmask on address), hoisted out of K-loop
    const __bf16* a0 = ((sbase + ml + sh) >= 0)
                         ? (A + (size_t)(m0 + ml + sh)      * D_ + ko)
                         : (Zpage + ko);
    const __bf16* a1 = ((sbase + 16 + ml + sh) >= 0)
                         ? (A + (size_t)(m0 + 16 + ml + sh) * D_ + ko)
                         : (Zpage + ko);
    const __bf16* b0 = Wt + (size_t)t * D_ * D_ + (size_t)(n0 + ml) * D_ + kb;
    for (int k0 = 0; k0 < D_; k0 += 32) {
      v16b af0 = load_a(a0 + k0);
      v16b af1 = load_a(a1 + k0);
#pragma unroll
      for (int j = 0; j < 4; ++j) {
        v16b bf = *(const v16b*)(b0 + (size_t)j * 16 * D_ + k0);
        acc[0][j] = wmma_bf16(af0, bf, acc[0][j]);
        acc[1][j] = wmma_bf16(af1, bf, acc[1][j]);
      }
    }
  }

#pragma unroll
  for (int j = 0; j < 4; ++j) {
    const int c  = n0 + j * 16 + ml;
    const float bv = bias[c];
#pragma unroll
    for (int i = 0; i < 2; ++i)
#pragma unroll
      for (int r = 0; r < 8; ++r) {
        const int mr = m0 + i * 16 + hi * 8 + r;
        float v = acc[i][j][r] + bv;
        if (RES) v += (float)Res[(size_t)mr * D_ + c];
        if (RELU) v = fmaxf(v, 0.0f);
        C[(size_t)mr * D_ + c] = (OutT)v;
      }
  }
}

// ---------------- element-wise / prep kernels ----------------

__global__ void zero_bf16_k(__bf16* __restrict__ p, int n)
{
  int id = blockIdx.x * blockDim.x + threadIdx.x;
  if (id < n) p[id] = (__bf16)0.0f;
}

// conv weights: in [taps][Din][Dout] -> out [taps][Dout][Din] (bf16)
__global__ void transpose_cvt_k(const float* __restrict__ in,
                                __bf16* __restrict__ out, int taps)
{
  size_t id = (size_t)blockIdx.x * blockDim.x + threadIdx.x;
  size_t total = (size_t)taps * D_ * D_;
  if (id >= total) return;
  int i = (int)(id % D_);
  int o = (int)((id / D_) % D_);
  int t = (int)(id / ((size_t)D_ * D_));
  out[id] = (__bf16)in[(size_t)t * D_ * D_ + (size_t)i * D_ + o];
}

__global__ void cvt_k(const float* __restrict__ in, __bf16* __restrict__ out, size_t n)
{
  size_t id = (size_t)blockIdx.x * blockDim.x + threadIdx.x;
  if (id < n) out[id] = (__bf16)in[id];
}

// x = embed[x_in] * mask  -> bf16 [BS,D]
__global__ void gather_embed_k(const int* __restrict__ xin,
                               const unsigned char* __restrict__ mask,
                               const float* __restrict__ embed,
                               __bf16* __restrict__ xb)
{
  size_t id = (size_t)blockIdx.x * blockDim.x + threadIdx.x; // BS * D/4 threads
  const int r  = (int)(id / (D_ / 4));
  const int d4 = (int)(id % (D_ / 4));
  const int idx = xin[r];
  const float mk = mask[r] ? 1.0f : 0.0f;
  const float4 e = *((const float4*)(embed + (size_t)idx * D_) + d4);
  v4b o;
  o[0] = (__bf16)(e.x * mk); o[1] = (__bf16)(e.y * mk);
  o[2] = (__bf16)(e.z * mk); o[3] = (__bf16)(e.w * mk);
  *((v4b*)(xb + (size_t)r * D_) + d4) = o;
}

// X = embed[sub_nodes] -> bf16 [N,D]
__global__ void gather_sub_k(const int* __restrict__ nodes,
                             const float* __restrict__ embed,
                             __bf16* __restrict__ Xb)
{
  size_t id = (size_t)blockIdx.x * blockDim.x + threadIdx.x; // N * D/4 threads
  const int r  = (int)(id / (D_ / 4));
  const int d4 = (int)(id % (D_ / 4));
  const int idx = nodes[r];
  const float4 e = *((const float4*)(embed + (size_t)idx * D_) + d4);
  v4b o;
  o[0] = (__bf16)e.x; o[1] = (__bf16)e.y; o[2] = (__bf16)e.z; o[3] = (__bf16)e.w;
  *((v4b*)(Xb + (size_t)r * D_) + d4) = o;
}

// LayerNorm over last dim (512) of (conv2+res) f32, * mask -> bf16
__global__ __launch_bounds__(256) void ln_k(
    const float* __restrict__ T, const float* __restrict__ g,
    const float* __restrict__ bb, const unsigned char* __restrict__ mask,
    __bf16* __restrict__ Y)
{
  const int row  = blockIdx.x * 8 + threadIdx.y;  // wave per row
  const int lane = threadIdx.x;
  const float* t = T + (size_t)row * D_;
  float vals[16];
  float s = 0.0f, ss = 0.0f;
#pragma unroll
  for (int i = 0; i < 16; ++i) {
    float v = t[lane + 32 * i];
    vals[i] = v; s += v; ss += v * v;
  }
#pragma unroll
  for (int off = 16; off > 0; off >>= 1) {
    s  += __shfl_xor(s,  off, 32);
    ss += __shfl_xor(ss, off, 32);
  }
  const float mu  = s * (1.0f / D_);
  const float var = ss * (1.0f / D_) - mu * mu;
  const float rs  = rsqrtf(var + 1e-5f);
  const float mk  = mask[row] ? 1.0f : 0.0f;
#pragma unroll
  for (int i = 0; i < 16; ++i) {
    const int d = lane + 32 * i;
    float v = ((vals[i] - mu) * rs * g[d] + bb[d]) * mk;
    Y[(size_t)row * D_ + d] = (__bf16)(v);
  }
}

// inv_sqrt[i] = clip(1 + sum_j A[i][j], 1e-6)^-0.5    (wave per row, N=2048)
__global__ __launch_bounds__(256) void degree_k(const float* __restrict__ A,
                                                float* __restrict__ isq)
{
  const int row  = blockIdx.x * 8 + threadIdx.y;
  const int lane = threadIdx.x;
  const float* a = A + (size_t)row * N_;
  float s = 0.0f;
#pragma unroll 4
  for (int i = 0; i < N_ / 32; ++i) s += a[lane + 32 * i];
#pragma unroll
  for (int off = 16; off > 0; off >>= 1) s += __shfl_xor(s, off, 32);
  s = fmaxf(s + 1.0f, 1e-6f);
  if (lane == 0) isq[row] = rsqrtf(s);
}

// A_norm[i][j] = isq[i]*(A[i][j] + (i==j))*isq[j]  -> bf16
__global__ void anorm_k(const float* __restrict__ A, const float* __restrict__ isq,
                        __bf16* __restrict__ An)
{
  size_t id = (size_t)blockIdx.x * blockDim.x + threadIdx.x;
  if (id >= (size_t)N_ * N_) return;
  const int i = (int)(id / N_);
  const int j = (int)(id % N_);
  float v = A[id] + (i == j ? 1.0f : 0.0f);
  An[id] = (__bf16)(isq[i] * v * isq[j]);
}

// =====================================================================
extern "C" void kernel_launch(void* const* d_in, const int* in_sizes, int n_in,
                              void* d_out, int out_size, void* d_ws, size_t ws_size,
                              hipStream_t stream)
{
  (void)in_sizes; (void)n_in; (void)out_size; (void)ws_size;

  const int*           x_in      = (const int*)d_in[0];
  const unsigned char* mask      = (const unsigned char*)d_in[1];
  const int*           sub_nodes = (const int*)d_in[2];
  const float*         A_sub     = (const float*)d_in[3];
  const float*         embed     = (const float*)d_in[4];
  const float*         conv1_w   = (const float*)d_in[5];
  const float*         conv1_b   = (const float*)d_in[6];
  const float*         conv2_w   = (const float*)d_in[7];
  const float*         conv2_b   = (const float*)d_in[8];
  const float*         ln_g      = (const float*)d_in[9];
  const float*         ln_b      = (const float*)d_in[10];
  const float*         gcn_w     = (const float*)d_in[11];
  const float*         gcn_b     = (const float*)d_in[12];
  const float*         hproj_w   = (const float*)d_in[13];
  float*               out       = (float*)d_out;

  // -------- carve workspace (aligned 256B); peak ~150 MB --------
  char* p = (char*)d_ws;
  auto carve = [&](size_t bytes) -> void* {
    void* r = (void*)p;
    p += (bytes + 255) & ~(size_t)255;
    return r;
  };
  __bf16* xb  = (__bf16*)carve((size_t)BS_ * D_ * 2);  // x (bf16); later reused as h
  __bf16* y1  = (__bf16*)carve((size_t)BS_ * D_ * 2);  // conv1 out; later reused as y
  float*  tmp = (float*) carve((size_t)BS_ * D_ * 4);  // conv2 out + residual (f32)
  __bf16* w1T = (__bf16*)carve((size_t)3 * D_ * D_ * 2);
  __bf16* w2T = (__bf16*)carve((size_t)3 * D_ * D_ * 2);
  __bf16* hpb = (__bf16*)carve((size_t)D_ * D_ * 2);
  __bf16* gwb = (__bf16*)carve((size_t)D_ * D_ * 2);
  __bf16* Xb  = (__bf16*)carve((size_t)N_ * D_ * 2);
  __bf16* XWt = (__bf16*)carve((size_t)N_ * D_ * 2);   // [D][N] (transposed)
  __bf16* Hb  = (__bf16*)carve((size_t)N_ * D_ * 2);
  __bf16* An  = (__bf16*)carve((size_t)N_ * N_ * 2);
  float*  isq = (float*) carve((size_t)N_ * 4);
  __bf16* zpg = (__bf16*)carve((size_t)ZPAD_ * 2);     // zero page for causal pad
  __bf16* yb  = y1;   // LN output reuses y1 (dead after conv2)
  __bf16* hb  = xb;   // hproj output reuses xb (dead after conv2 residual)

  const dim3 b128(32, 4);
  const dim3 b256(32, 8);

  // -------- prep: zero page + weights (bf16, conv weights -> [t][o][i]) -----
  zero_bf16_k<<<(ZPAD_ + 255) / 256, 256, 0, stream>>>(zpg, ZPAD_);
  const int twg = (3 * D_ * D_) / 256;
  transpose_cvt_k<<<twg, 256, 0, stream>>>(conv1_w, w1T, 3);
  transpose_cvt_k<<<twg, 256, 0, stream>>>(conv2_w, w2T, 3);
  cvt_k<<<(D_ * D_) / 256, 256, 0, stream>>>(hproj_w, hpb, (size_t)D_ * D_);
  cvt_k<<<(D_ * D_) / 256, 256, 0, stream>>>(gcn_w, gwb, (size_t)D_ * D_);

  // -------- encode path --------
  gather_embed_k<<<(BS_ * (D_ / 4)) / 256, 256, 0, stream>>>(x_in, mask, embed, xb);

  conv_k<true, false, __bf16><<<dim3(D_ / 256, BS_ / 32), b128, 0, stream>>>(
      xb, w1T, y1, conv1_b, nullptr, zpg);                  // y1 = relu(conv1(x))

  conv_k<false, true, float><<<dim3(D_ / 256, BS_ / 32), b128, 0, stream>>>(
      y1, w2T, tmp, conv2_b, xb, zpg);                      // tmp = conv2(y1) + x

  ln_k<<<BS_ / 8, b256, 0, stream>>>(tmp, ln_g, ln_b, mask, yb);  // yb = LN(tmp)*mask

  gemm_k<__bf16, false, false><<<dim3(D_ / 256, BS_ / 32), b128, 0, stream>>>(
      yb, hpb, hb, nullptr, D_, D_);                        // h = y @ hproj^T

  // -------- GCN branch --------
  degree_k<<<N_ / 8, b256, 0, stream>>>(A_sub, isq);
  anorm_k<<<(int)(((size_t)N_ * N_) / 256), 256, 0, stream>>>(A_sub, isq, An);
  gather_sub_k<<<(N_ * (D_ / 4)) / 256, 256, 0, stream>>>(sub_nodes, embed, Xb);

  gemm_k<__bf16, false, true><<<dim3(D_ / 256, N_ / 32), b128, 0, stream>>>(
      Xb, gwb, XWt, nullptr, D_, N_);                       // XWt[d][n] = (X@gcn_w^T)^T

  gemm_k<__bf16, true, false><<<dim3(D_ / 256, N_ / 32), b128, 0, stream>>>(
      An, XWt, Hb, gcn_b, N_, D_);                          // H = relu(A_norm@XW + b)

  // -------- logits = h @ H^T  (writes d_out directly) --------
  gemm_k<float, false, false><<<dim3(N_ / 256, BS_ / 32), b128, 0, stream>>>(
      hb, Hb, out, nullptr, D_, N_);
}
